// YvFlashSSM_72112500900651
// MI455X (gfx1250) — matrix operand order repaired and checked by reading, not compiled, not run on hardware
//
#include <hip/hip_runtime.h>
#include <hip/hip_bf16.h>

typedef __attribute__((ext_vector_type(16))) __bf16 v16bf;
typedef __attribute__((ext_vector_type(8)))  __bf16 v8bf;
typedef __attribute__((ext_vector_type(8)))  float  v8f;

#define D_MODEL 1024
#define SEQ     2048
#define BATCH   2
#define TOKENS  (BATCH * SEQ)   // 4096
#define NSTATE  32

// ---- gfx1250 async global->LDS copy (ASYNCcnt), with safe fallbacks ------
// clang signature (from diagnostics): first param is
//   __attribute__((vector_size(16))) int __device__*  (global, AS1)
// second is the LDS destination (AS3), then two immediate ints.
typedef int v4i_async __attribute__((vector_size(16)));

#if __has_builtin(__builtin_amdgcn_global_load_async_to_lds_b128)
#define HAVE_ASYNC_LDS 1
#define ASYNC_CP_B128(gsrc, ldst)                                            \
  __builtin_amdgcn_global_load_async_to_lds_b128(                            \
      (__attribute__((address_space(1))) v4i_async*)(gsrc),                  \
      (__attribute__((address_space(3))) v4i_async*)(ldst), 0, 0)
#if __has_builtin(__builtin_amdgcn_s_wait_asynccnt)
#define ASYNC_WAIT() __builtin_amdgcn_s_wait_asynccnt(0)
#else
#define ASYNC_WAIT() asm volatile("s_wait_asynccnt 0x0" ::: "memory")
#endif
#else
#define HAVE_ASYNC_LDS 0
#define ASYNC_WAIT()
#endif

// ---------------------------------------------------------------------------
// GEMM: C[M x N] = A[M x K] * W^T, W is [N x K] row-major, K = 1024.
// bf16 inputs (converted while staging), f32 accumulate via
// v_wmma_f32_16x16x32_bf16. Block tile 128M x 128N, K-slab 32, 8 waves,
// wave tile 32M x 64N (2x4 subtiles). Double-buffered LDS + register
// pipelining: global loads for slab k+1 are in flight during slab k's WMMAs.
//
// MODE 0: A = x row-major (token, k). Epilogue splits columns:
//         n<1024 -> u_raw ; 1024..2047 -> softplus -> delta ; 2048..2111 -> BC
// MODE 1: A = z_T (b, d, t) transposed layout. Epilogue: plain store.
// ---------------------------------------------------------------------------
template <int MODE>
__global__ __launch_bounds__(256)
void ssm_gemm_kernel(const float* __restrict__ Ag,
                     const float* __restrict__ Wg,
                     float* __restrict__ out_u,
                     float* __restrict__ out_delta,
                     float* __restrict__ out_bc,
                     float* __restrict__ out_plain) {
  __shared__ __align__(16) __bf16 Alds[2][128][40];  // 32 k + 8 pad
  __shared__ __align__(16) __bf16 Blds[2][128][40];

  const int tid  = threadIdx.x;
  const int lane = tid & 31;
  const int wave = tid >> 5;
  const int wr = wave >> 1;     // 0..3  (M)
  const int wc = wave & 1;      // 0..1  (N)

  const int ntile  = blockIdx.x;
  const int mtile  = blockIdx.y;
  const int token0 = mtile * 128;
  const int n0     = ntile * 128;

  const int KITERS = D_MODEL / 32;
  const int b_of_tile = token0 >> 11;     // MODE 1 only
  const int t_of_tile = token0 & (SEQ - 1);

  v8f acc[2][4];
#pragma unroll
  for (int i = 0; i < 2; ++i)
#pragma unroll
    for (int j = 0; j < 4; ++j) acc[i][j] = (v8f)0.0f;

  auto loadA = [&](int kt, float4* r) {
    if (MODE == 0) {
      const float* src =
          Ag + (size_t)(token0 + (tid >> 1)) * D_MODEL + kt * 32 + (tid & 1) * 16;
#pragma unroll
      for (int q = 0; q < 4; ++q) r[q] = ((const float4*)src)[q];
    } else {
      const float* src = Ag +
          (size_t)((b_of_tile << 10) + kt * 32 + (tid >> 3)) * SEQ +
          t_of_tile + (tid & 7) * 16;
#pragma unroll
      for (int q = 0; q < 4; ++q) r[q] = ((const float4*)src)[q];
    }
  };
  auto loadB = [&](int kt, float4* r) {
    const float* src =
        Wg + (size_t)(n0 + (tid >> 1)) * D_MODEL + kt * 32 + (tid & 1) * 16;
#pragma unroll
    for (int q = 0; q < 4; ++q) r[q] = ((const float4*)src)[q];
  };
  auto storeA = [&](int buf, const float4* r) {
    if (MODE == 0) {
      const int row = tid >> 1, col = (tid & 1) * 16;
#pragma unroll
      for (int q = 0; q < 4; ++q) {
        Alds[buf][row][col + q * 4 + 0] = (__bf16)r[q].x;
        Alds[buf][row][col + q * 4 + 1] = (__bf16)r[q].y;
        Alds[buf][row][col + q * 4 + 2] = (__bf16)r[q].z;
        Alds[buf][row][col + q * 4 + 3] = (__bf16)r[q].w;
      }
    } else {
      const int k = tid >> 3, m0 = (tid & 7) * 16;
#pragma unroll
      for (int q = 0; q < 4; ++q) {
        Alds[buf][m0 + q * 4 + 0][k] = (__bf16)r[q].x;
        Alds[buf][m0 + q * 4 + 1][k] = (__bf16)r[q].y;
        Alds[buf][m0 + q * 4 + 2][k] = (__bf16)r[q].z;
        Alds[buf][m0 + q * 4 + 3][k] = (__bf16)r[q].w;
      }
    }
  };
  auto storeB = [&](int buf, const float4* r) {
    const int n = tid >> 1, col = (tid & 1) * 16;
#pragma unroll
    for (int q = 0; q < 4; ++q) {
      Blds[buf][n][col + q * 4 + 0] = (__bf16)r[q].x;
      Blds[buf][n][col + q * 4 + 1] = (__bf16)r[q].y;
      Blds[buf][n][col + q * 4 + 2] = (__bf16)r[q].z;
      Blds[buf][n][col + q * 4 + 3] = (__bf16)r[q].w;
    }
  };

  float4 ra[4], rb[4];
  loadA(0, ra);
  loadB(0, rb);

  const int half = lane >> 4;   // 0 / 1
  const int lm   = lane & 15;

  for (int kt = 0; kt < KITERS; ++kt) {
    const int buf = kt & 1;
    storeA(buf, ra);
    storeB(buf, rb);
    __syncthreads();
    if (kt + 1 < KITERS) {      // issue next slab's loads before computing
      loadA(kt + 1, ra);
      loadB(kt + 1, rb);
    }

    v16bf afrag[2], bfrag[4];
#pragma unroll
    for (int ms = 0; ms < 2; ++ms) {
      const __bf16* arow = &Alds[buf][wr * 32 + ms * 16 + lm][0];
      v8bf lo = *(const v8bf*)(arow + half * 8);        // K = 8*half + 0..7
      v8bf hi = *(const v8bf*)(arow + 16 + half * 8);   // K = 16+8*half+0..7
      afrag[ms] = __builtin_shufflevector(lo, hi, 0, 1, 2, 3, 4, 5, 6, 7,
                                          8, 9, 10, 11, 12, 13, 14, 15);
    }
#pragma unroll
    for (int ns = 0; ns < 4; ++ns) {
      const __bf16* brow = &Blds[buf][wc * 64 + ns * 16 + lm][0];
      v8bf lo = *(const v8bf*)(brow + half * 8);        // K = 8*half + 0..7
      v8bf hi = *(const v8bf*)(brow + 16 + half * 8);   // K = 16+8*half+0..7
      bfrag[ns] = __builtin_shufflevector(lo, hi, 0, 1, 2, 3, 4, 5, 6, 7,
                                          8, 9, 10, 11, 12, 13, 14, 15);
    }
#pragma unroll
    for (int ms = 0; ms < 2; ++ms)
#pragma unroll
      for (int ns = 0; ns < 4; ++ns)
        acc[ms][ns] = __builtin_amdgcn_wmma_f32_16x16x32_bf16(
            false, afrag[ms], false, bfrag[ns], (short)0, acc[ms][ns],
            false, false);
  }

  // ---- epilogue ----
#pragma unroll
  for (int ms = 0; ms < 2; ++ms) {
#pragma unroll
    for (int ns = 0; ns < 4; ++ns) {
#pragma unroll
      for (int r = 0; r < 8; ++r) {
        const int token = token0 + wr * 32 + ms * 16 + r + half * 8;
        const int n     = n0 + wc * 64 + ns * 16 + lm;
        const float val = acc[ms][ns][r];
        if (MODE == 0) {
          if (n < D_MODEL) {
            out_u[(size_t)token * D_MODEL + n] = val;
          } else if (n < 2 * D_MODEL) {
            float sp = (val > 20.0f) ? val : log1pf(__expf(val));
            out_delta[(size_t)token * D_MODEL + (n - D_MODEL)] = sp;
          } else if (n < 2 * D_MODEL + 2 * NSTATE) {
            out_bc[(size_t)token * (2 * NSTATE) + (n - 2 * D_MODEL)] = val;
          }
        } else {
          out_plain[(size_t)token * D_MODEL + n] = val;
        }
      }
    }
  }
}

// ---------------------------------------------------------------------------
// Depthwise causal conv (K=4) + bias on u, and transpose u & delta to
// (b, d, t) layout for contiguous scan streaming. Tile: 64 t x 64 d.
// Tile staging uses async global->LDS B128 copies (raw copies, no convert).
// ---------------------------------------------------------------------------
__global__ __launch_bounds__(256)
void conv_transpose_kernel(const float* __restrict__ u_raw,
                           const float* __restrict__ delta,
                           const float* __restrict__ conv_w,
                           const float* __restrict__ conv_b,
                           float* __restrict__ u_T,
                           float* __restrict__ d_T) {
  __shared__ __align__(16) float su[67][68];  // rows t0-3 .. t0+63 (16B rows)
  __shared__ float so[64][67];                // conv out, transpose-read

  const int tid = threadIdx.x;
  const int token0 = blockIdx.x * 64;
  const int d0 = blockIdx.y * 64;
  const int b  = token0 >> 11;
  const int t0 = token0 & (SEQ - 1);

  // ---- stage u rows [t0-3, t0+63] x [d0, d0+63], zero-pad t<0 ----
  for (int i = tid; i < 67 * 16; i += 256) {
    const int row = i >> 4;
    const int c4  = (i & 15) * 4;
    const int t   = t0 - 3 + row;
#if HAVE_ASYNC_LDS
    if (t >= 0) {
      ASYNC_CP_B128(u_raw + (size_t)(b * SEQ + t) * D_MODEL + d0 + c4,
                    &su[row][c4]);
    } else {
      su[row][c4 + 0] = 0.0f; su[row][c4 + 1] = 0.0f;
      su[row][c4 + 2] = 0.0f; su[row][c4 + 3] = 0.0f;
    }
#else
    float4 v = make_float4(0.f, 0.f, 0.f, 0.f);
    if (t >= 0)
      v = *(const float4*)(u_raw + (size_t)(b * SEQ + t) * D_MODEL + d0 + c4);
    su[row][c4 + 0] = v.x; su[row][c4 + 1] = v.y;
    su[row][c4 + 2] = v.z; su[row][c4 + 3] = v.w;
#endif
  }
  ASYNC_WAIT();
  __syncthreads();

  // ---- depthwise conv along t ----
  {
    const int c = tid & 63;
    const int rg = tid >> 6;                // 0..3
    const float w0 = conv_w[(d0 + c) * 4 + 0];
    const float w1 = conv_w[(d0 + c) * 4 + 1];
    const float w2 = conv_w[(d0 + c) * 4 + 2];
    const float w3 = conv_w[(d0 + c) * 4 + 3];
    const float cb = conv_b[d0 + c];
#pragma unroll
    for (int i = 0; i < 16; ++i) {
      const int t = rg * 16 + i;
      so[t][c] = cb + w0 * su[t][c] + w1 * su[t + 1][c] + w2 * su[t + 2][c] +
                 w3 * su[t + 3][c];
    }
  }
  __syncthreads();

  // ---- write u_T (b,d,t) contiguous-in-t ; async-stage delta tile ----
  {
    const int dl = tid >> 2, tseg = tid & 3;
    float* dst = u_T + (size_t)((b << 10) + d0 + dl) * SEQ + t0 + tseg * 16;
#pragma unroll
    for (int q = 0; q < 4; ++q) {
      float4 v;
      v.x = so[tseg * 16 + q * 4 + 0][dl];
      v.y = so[tseg * 16 + q * 4 + 1][dl];
      v.z = so[tseg * 16 + q * 4 + 2][dl];
      v.w = so[tseg * 16 + q * 4 + 3][dl];
      *(float4*)(dst + q * 4) = v;
    }
    for (int i = tid; i < 64 * 16; i += 256) {
      const int row = i >> 4;
      const int c4  = (i & 15) * 4;
#if HAVE_ASYNC_LDS
      ASYNC_CP_B128(delta + (size_t)(b * SEQ + t0 + row) * D_MODEL + d0 + c4,
                    &su[row][c4]);
#else
      float4 v =
          *(const float4*)(delta + (size_t)(b * SEQ + t0 + row) * D_MODEL +
                           d0 + c4);
      su[row][c4 + 0] = v.x; su[row][c4 + 1] = v.y;
      su[row][c4 + 2] = v.z; su[row][c4 + 3] = v.w;
#endif
    }
  }
  ASYNC_WAIT();
  __syncthreads();

  // ---- write delta_T ----
  {
    const int dl = tid >> 2, tseg = tid & 3;
    float* dst = d_T + (size_t)((b << 10) + d0 + dl) * SEQ + t0 + tseg * 16;
#pragma unroll
    for (int q = 0; q < 4; ++q) {
      float4 v;
      v.x = su[tseg * 16 + q * 4 + 0][dl];
      v.y = su[tseg * 16 + q * 4 + 1][dl];
      v.z = su[tseg * 16 + q * 4 + 2][dl];
      v.w = su[tseg * 16 + q * 4 + 3][dl];
      *(float4*)(dst + q * 4) = v;
    }
  }
}

// ---------------------------------------------------------------------------
// Selective scan: one wave32 per (b,d) channel, lane = state index n (N=32).
//   h = h*exp(delta_t * A[d,n]) + B_t[n]*u_t ;  y_t = sum_n h*C_t[n]
//   z_t = y_t + u_t * D[d]  (fused skip), stored transposed (b,d,t).
// B/C staged per 32-t chunk via async global->LDS, shared by the block's
// 8 waves (all same batch b). Butterfly reduction via __shfl_xor (wave32).
// ---------------------------------------------------------------------------
__global__ __launch_bounds__(256)
void ssm_scan_kernel(const float* __restrict__ d_T,
                     const float* __restrict__ u_T,
                     const float* __restrict__ bc,
                     const float* __restrict__ A_log,
                     const float* __restrict__ D_param,
                     float* __restrict__ z_T) {
  __shared__ __align__(16) float sbc[32 * 64];

  const int lane = threadIdx.x & 31;
  const int wave = threadIdx.x >> 5;
  const int w = blockIdx.x * 8 + wave;     // 0..2047
  const int b = w >> 10;
  const int d = w & (D_MODEL - 1);

  const float A  = -__expf(A_log[lane * D_MODEL + d]);  // A_T[d, lane]
  const float Dp = D_param[d];
  const size_t chbase = (size_t)((b << 10) + d) * SEQ;

  float h = 0.0f;
  for (int t0 = 0; t0 < SEQ; t0 += 32) {
    // stage BC chunk: 32 t x 64 floats = 8KB
    {
      const float* src = bc + (size_t)(b * SEQ + t0) * 64 + threadIdx.x * 8;
#if HAVE_ASYNC_LDS
      ASYNC_CP_B128(src, &sbc[threadIdx.x * 8]);
      ASYNC_CP_B128(src + 4, &sbc[threadIdx.x * 8 + 4]);
#else
#pragma unroll
      for (int q = 0; q < 8; ++q) sbc[threadIdx.x * 8 + q] = src[q];
#endif
    }
    ASYNC_WAIT();
    __syncthreads();

    const float dl = d_T[chbase + t0 + lane];   // delta (softplus'd)
    const float uu = u_T[chbase + t0 + lane];   // conv'd u
    float zacc = 0.0f;

#pragma unroll
    for (int j = 0; j < 32; ++j) {
      const float dj = __shfl(dl, j, 32);
      const float uj = __shfl(uu, j, 32);
      const float Bn = sbc[j * 64 + lane];
      const float Cn = sbc[j * 64 + 32 + lane];
      h = h * __expf(dj * A) + Bn * uj;
      float p = h * Cn;
      p += __shfl_xor(p, 16, 32);
      p += __shfl_xor(p, 8, 32);
      p += __shfl_xor(p, 4, 32);
      p += __shfl_xor(p, 2, 32);
      p += __shfl_xor(p, 1, 32);
      const float zj = p + uj * Dp;
      zacc = (lane == j) ? zj : zacc;
    }
    z_T[chbase + t0 + lane] = zacc;
    __syncthreads();
  }
}

// ---------------------------------------------------------------------------
extern "C" void kernel_launch(void* const* d_in, const int* in_sizes, int n_in,
                              void* d_out, int out_size, void* d_ws,
                              size_t ws_size, hipStream_t stream) {
  const float* x       = (const float*)d_in[0];
  const float* W_in    = (const float*)d_in[1];
  const float* W_out   = (const float*)d_in[2];
  const float* conv_w  = (const float*)d_in[3];
  const float* conv_b  = (const float*)d_in[4];
  const float* A_log   = (const float*)d_in[5];
  const float* D_param = (const float*)d_in[6];
  float* out = (float*)d_out;

  float* ws = (float*)d_ws;
  const size_t PLANE = (size_t)TOKENS * D_MODEL;   // 4M floats
  float* u_raw = ws;
  float* delta = ws + PLANE;
  float* bc    = ws + 2 * PLANE;
  float* u_T   = bc + (size_t)TOKENS * 64;
  float* dl_T  = u_T + PLANE;
  float* z_T   = dl_T + PLANE;

  // 1) input projection (2112 useful cols), fused softplus, split stores
  ssm_gemm_kernel<0><<<dim3(17, TOKENS / 128), 256, 0, stream>>>(
      x, W_in, u_raw, delta, bc, nullptr);

  // 2) depthwise causal conv + bias; transpose u and delta to (b,d,t)
  conv_transpose_kernel<<<dim3(TOKENS / 64, D_MODEL / 64), 256, 0, stream>>>(
      u_raw, delta, conv_w, conv_b, u_T, dl_T);

  // 3) selective scan (wave32 = 32 states), fused +u*D skip, writes z_T
  ssm_scan_kernel<<<dim3(256), 256, 0, stream>>>(dl_T, u_T, bc, A_log,
                                                 D_param, z_T);

  // 4) output projection from transposed z
  ssm_gemm_kernel<1><<<dim3(D_MODEL / 128, TOKENS / 128), 256, 0, stream>>>(
      z_T, W_out, nullptr, nullptr, nullptr, out);
}
